// MultiHeadAttention_79723182948800
// MI455X (gfx1250) — matrix operand order repaired and checked by reading, not compiled, not run on hardware
//
#include <hip/hip_runtime.h>
#include <hip/hip_bf16.h>
#include <cstdint>
#include <cstddef>

// MHA forward for B=2, S=2048, D=1024, H=16, HD=64 on gfx1250 (wave32, WMMA).
// f16 WMMA with f32 accumulation; flash-attention causal softmax; async
// global->LDS staging (ASYNCcnt) for GEMM tiles and attention K tiles.

typedef __attribute__((ext_vector_type(16))) _Float16 v16h;
typedef __attribute__((ext_vector_type(8)))  float    v8f;

#define B_  2
#define S_  2048
#define D_  1024
#define H_  16
#define HD_ 64

// Async copy: 16 bytes global -> LDS per active lane, tracked by ASYNCcnt.
// LDS operand is the low 32 bits of the generic pointer (ISA 10.2), for
// shared-aperture addresses.
#if defined(__gfx1250__)
#define ASYNC_LDS_B128(ldsOff, gptr)                                        \
  asm volatile("global_load_async_to_lds_b128 %0, %1, off"                  \
               :: "v"(ldsOff), "v"(gptr) : "memory")
#define ASYNC_WAIT_ALL() asm volatile("s_wait_asynccnt 0" ::: "memory")
#define USE_ASYNC_LDS 1
#else
#define ASYNC_LDS_B128(ldsOff, gptr) (void)0
#define ASYNC_WAIT_ALL() (void)0
#define USE_ASYNC_LDS 0
#endif

// ---------------------------------------------------------------------------
// Fragment gathers following CDNA5 ISA 7.12.2 (16-bit A 16x32 / B 32x16).
// A (16xK tile, row-major in LDS, rowStrideDw dwords per row):
//   lane m=l&15, kh=l>>4;  VGPR v holds K = (v<4 ? 2v : 16+2(v-4)) + 8*kh, +1
// B (KxN tile; stored as N rows of K halves, row-major):
//   lane n=l&15, kh=l>>4;  VGPR v holds K = (v<4 ? 2v : 16+2(v-4)) + 8*kh, +1
// Both patterns are dword-gathers that the compiler fuses into ds_load_b128.
// ---------------------------------------------------------------------------
static __device__ __forceinline__ v16h frag_a_from(const uint32_t* base, int rowStrideDw) {
  const int lane = threadIdx.x & 31;
  const int m = lane & 15, kh = lane >> 4;
  union { v16h h; uint32_t u[8]; } f;
#pragma unroll
  for (int v = 0; v < 8; ++v) {
    const int kbase = ((v < 4) ? (2 * v) : (16 + 2 * (v - 4))) + 8 * kh;
    f.u[v] = base[m * rowStrideDw + (kbase >> 1)];
  }
  return f.h;
}

static __device__ __forceinline__ v16h frag_b_from(const uint32_t* base, int rowStrideDw) {
  const int lane = threadIdx.x & 31;
  const int n = lane & 15, kh = lane >> 4;
  union { v16h h; uint32_t u[8]; } f;
#pragma unroll
  for (int v = 0; v < 8; ++v) {
    const int kbase = ((v < 4) ? (2 * v) : (16 + 2 * (v - 4))) + 8 * kh;
    f.u[v] = base[n * rowStrideDw + (kbase >> 1)];
  }
  return f.h;
}

// ---------------------------------------------------------------------------
// Elementwise fp32 -> f16
// ---------------------------------------------------------------------------
__global__ void cvt_f32_to_f16(const float* __restrict__ in, _Float16* __restrict__ out, int n) {
  int i = blockIdx.x * blockDim.x + threadIdx.x;
  const int stride = gridDim.x * blockDim.x;
  for (; i < n; i += stride) out[i] = (_Float16)in[i];
}

// ---------------------------------------------------------------------------
// Y[M,N] = X[M,K] @ W[N,K]^T + bias[N]   (torch Linear)
// Block tile 64x128, 4 waves each computing a 32x64 quadrant (8 WMMA tiles).
// K-step 64, double-buffered LDS (2 x (64x64 A + 128x64 B) f16 = 48 KB).
// Per wave per K-step: 4 A-frags + 8 B-frags -> 16 WMMAs.
// ---------------------------------------------------------------------------
template <bool OUT_F32>
__global__ __launch_bounds__(128)
void gemm_xwT_bias(const _Float16* __restrict__ X,
                   const _Float16* __restrict__ W,
                   const float*    __restrict__ bias,
                   _Float16* __restrict__ Yh,
                   float*    __restrict__ Yf,
                   int M, int N, int K) {
  __shared__ _Float16 As[2][64 * 64];    // [buf][row][k]  row stride 64 halves
  __shared__ _Float16 Bs[2][128 * 64];

  const int t    = threadIdx.x;
  const int wave = t >> 5;
  const int lane = t & 31;
  const int m0 = blockIdx.y * 64;
  const int n0 = blockIdx.x * 128;
  const int wm = (wave >> 1) * 32;  // wave's M offset within tile
  const int wn = (wave & 1) * 64;   // wave's N offset within tile

  v8f acc[2][4];
#pragma unroll
  for (int i = 0; i < 2; ++i)
#pragma unroll
    for (int j = 0; j < 4; ++j)
#pragma unroll
      for (int r = 0; r < 8; ++r) acc[i][j][r] = 0.0f;

  const uint4* Xv = (const uint4*)X;   // 8 halves per uint4
  const uint4* Wv = (const uint4*)W;
  const int kv = K >> 3;               // uint4 per row
  const int nsteps = K >> 6;           // K / 64

  // Stage one 64-wide K slice into buffer `buf`.
  auto stage = [&](int s, int buf) {
    const int kk8 = s * 8;             // (s*64)/8, uint4 offset in a row
#if USE_ASYNC_LDS
#pragma unroll
    for (int i = 0; i < 4; ++i) {      // A: 64 rows x 8 uint4 = 512
      const int idx = t + i * 128;
      const int row = idx >> 3;
      const int c   = idx & 7;
      ASYNC_LDS_B128((uint32_t)(uintptr_t)&((uint4*)As[buf])[idx],
                     &Xv[(size_t)(m0 + row) * kv + kk8 + c]);
    }
#pragma unroll
    for (int i = 0; i < 8; ++i) {      // B: 128 rows x 8 uint4 = 1024
      const int idx = t + i * 128;
      const int row = idx >> 3;
      const int c   = idx & 7;
      ASYNC_LDS_B128((uint32_t)(uintptr_t)&((uint4*)Bs[buf])[idx],
                     &Wv[(size_t)(n0 + row) * kv + kk8 + c]);
    }
#else
#pragma unroll
    for (int i = 0; i < 4; ++i) {
      const int idx = t + i * 128;
      const int row = idx >> 3;
      const int c   = idx & 7;
      ((uint4*)As[buf])[idx] = Xv[(size_t)(m0 + row) * kv + kk8 + c];
    }
#pragma unroll
    for (int i = 0; i < 8; ++i) {
      const int idx = t + i * 128;
      const int row = idx >> 3;
      const int c   = idx & 7;
      ((uint4*)Bs[buf])[idx] = Wv[(size_t)(n0 + row) * kv + kk8 + c];
    }
#endif
  };

  stage(0, 0);
  ASYNC_WAIT_ALL();
  __syncthreads();

  for (int s = 0; s < nsteps; ++s) {
    const int cur = s & 1;
    if (s + 1 < nsteps) stage(s + 1, (s + 1) & 1);
    if (s + 2 < nsteps) {
      // Prime far caches two steps ahead (global_prefetch_b8).
      const int kk8 = (s + 2) * 8;
      __builtin_prefetch(&Xv[(size_t)(m0 + (t >> 3)) * kv + kk8 + (t & 7)], 0, 1);
      __builtin_prefetch(&Wv[(size_t)(n0 + (t >> 3)) * kv + kk8 + (t & 7)], 0, 1);
    }

#pragma unroll
    for (int kh2 = 0; kh2 < 2; ++kh2) {        // two K=32 halves of the 64 slice
      v16h a[2], b[4];
#pragma unroll
      for (int i = 0; i < 2; ++i)
        a[i] = frag_a_from((const uint32_t*)As[cur] + (wm + 16 * i) * 32 + kh2 * 16, 32);
#pragma unroll
      for (int j = 0; j < 4; ++j)
        b[j] = frag_b_from((const uint32_t*)Bs[cur] + (wn + 16 * j) * 32 + kh2 * 16, 32);
#pragma unroll
      for (int i = 0; i < 2; ++i)
#pragma unroll
        for (int j = 0; j < 4; ++j)
          acc[i][j] = __builtin_amdgcn_wmma_f32_16x16x32_f16(
              false, a[i], false, b[j], (short)0, acc[i][j], false, false);
    }
    ASYNC_WAIT_ALL();                  // next slice resident before the barrier
    __syncthreads();
  }

  // Epilogue. C/D layout: row M = r + 8*(lane>=16) in VGPR r; col N = lane&15.
  const int cn = lane & 15, kh = lane >> 4;
#pragma unroll
  for (int i = 0; i < 2; ++i)
#pragma unroll
    for (int j = 0; j < 4; ++j) {
      const int ncol = n0 + wn + 16 * j + cn;
      const float bvv = bias ? bias[ncol] : 0.0f;
#pragma unroll
      for (int r = 0; r < 8; ++r) {
        const int mrow = m0 + wm + 16 * i + r + 8 * kh;
        const float val = acc[i][j][r] + bvv;
        if (OUT_F32) Yf[(size_t)mrow * N + ncol] = val;
        else         Yh[(size_t)mrow * N + ncol] = (_Float16)val;
      }
    }
}

// ---------------------------------------------------------------------------
// Causal flash attention. Grid: (S/64, B*H). 4 waves; wave w owns query rows
// [qblock + 16w, +16). Keys staged 32 at a time (K tile async; V transposed).
// ---------------------------------------------------------------------------
__global__ __launch_bounds__(128)
void attn_flash(const _Float16* __restrict__ Qp,
                const _Float16* __restrict__ Kp,
                const _Float16* __restrict__ Vp,
                _Float16* __restrict__ Op) {
  __shared__ _Float16 Ks[32 * 64];      // [key][hd]
  __shared__ _Float16 Vt[64 * 32];      // [hd][key] (transposed)
  __shared__ _Float16 Ps[4][16 * 32];   // per-wave P scratch (re-swizzle D->A)

  const int t    = threadIdx.x;
  const int wave = t >> 5;
  const int lane = t & 31;
  const int bb = blockIdx.y / H_;
  const int h  = blockIdx.y % H_;
  const int qblock = blockIdx.x * 64;
  const int qbase  = qblock + wave * 16;
  const int cn = lane & 15, kh = lane >> 4;

  // Q fragments for hd 0..31 and 32..63, gathered straight from global.
  const uint32_t* gq = (const uint32_t*)Qp;
  v16h aq[2];
  {
    const size_t rowbase = ((size_t)(bb * S_ + qbase + cn) * D_ + h * HD_) >> 1;
#pragma unroll
    for (int half = 0; half < 2; ++half) {
      union { v16h h; uint32_t u[8]; } f;
#pragma unroll
      for (int v = 0; v < 8; ++v) {
        const int kbase = ((v < 4) ? (2 * v) : (16 + 2 * (v - 4))) + 8 * kh + 32 * half;
        f.u[v] = gq[rowbase + (kbase >> 1)];
      }
      aq[half] = f.h;
    }
  }

  v8f acc[4];
  float mrow[8], lrow[8];
#pragma unroll
  for (int j = 0; j < 4; ++j)
#pragma unroll
    for (int r = 0; r < 8; ++r) acc[j][r] = 0.0f;
#pragma unroll
  for (int r = 0; r < 8; ++r) { mrow[r] = -1e30f; lrow[r] = 0.0f; }

  const int kend = qblock + 64;  // causal bound for the whole block
  for (int kb = 0; kb < kend; kb += 32) {
    // Stage K tile [32 keys][64 hd] (256 uint4) via async copy.
    const uint4* Kv = (const uint4*)Kp;
#pragma unroll
    for (int i = 0; i < 2; ++i) {
      const int idx = t + i * 128;
      const int row = idx >> 3;        // 8 uint4 per key row
      const int c   = idx & 7;
#if USE_ASYNC_LDS
      ASYNC_LDS_B128((uint32_t)(uintptr_t)&((uint4*)Ks)[idx],
                     &Kv[((size_t)(bb * S_ + kb + row) * D_ + h * HD_) / 8 + c]);
#else
      ((uint4*)Ks)[idx] = Kv[((size_t)(bb * S_ + kb + row) * D_ + h * HD_) / 8 + c];
#endif
    }
    // Stage V transposed: Vt[hd][key] (scattered b16 stores; manual path).
    const uint32_t* Vv = (const uint32_t*)Vp;
#pragma unroll
    for (int i = 0; i < 8; ++i) {
      const int idx = t + i * 128;     // 1024 dwords, (key, hd-pair) order
      const int key = idx >> 5;        // 32 dwords per key row
      const int hd2 = idx & 31;
      const uint32_t d = Vv[((size_t)(bb * S_ + kb + key) * D_ + h * HD_) / 2 + hd2];
      union { uint32_t u; _Float16 hh[2]; } c2; c2.u = d;
      Vt[(2 * hd2 + 0) * 32 + key] = c2.hh[0];
      Vt[(2 * hd2 + 1) * 32 + key] = c2.hh[1];
    }
    ASYNC_WAIT_ALL();
    __syncthreads();

    if (kb <= qbase + 15) {            // some key in this tile is unmasked
      // S = Q K^T  (2 n-tiles of 16 keys; K-dim 64 = 2 WMMA steps each)
      v8f s[2];
#pragma unroll
      for (int tl = 0; tl < 2; ++tl) {
#pragma unroll
        for (int r = 0; r < 8; ++r) s[tl][r] = 0.0f;
#pragma unroll
        for (int hb = 0; hb < 2; ++hb) {
          v16h bk = frag_b_from((const uint32_t*)Ks + (16 * tl) * 32 + hb * 16, 32);
          s[tl] = __builtin_amdgcn_wmma_f32_16x16x32_f16(
              false, aq[hb], false, bk, (short)0, s[tl], false, false);
        }
      }
      // Online softmax per row (rows live in 16-lane halves; xor<16 reductions).
#pragma unroll
      for (int r = 0; r < 8; ++r) {
        const int q = qbase + r + 8 * kh;
        float v0 = s[0][r] * 0.125f;   // 1/sqrt(64)
        float v1 = s[1][r] * 0.125f;
        if (kb + cn      > q) v0 = -1e30f;
        if (kb + 16 + cn > q) v1 = -1e30f;
        float mx = fmaxf(v0, v1);
#pragma unroll
        for (int d = 1; d < 16; d <<= 1) mx = fmaxf(mx, __shfl_xor(mx, d, 32));
        const float newm = fmaxf(mrow[r], mx);
        const float p0 = __expf(v0 - newm);
        const float p1 = __expf(v1 - newm);
        float sum = p0 + p1;
#pragma unroll
        for (int d = 1; d < 16; d <<= 1) sum += __shfl_xor(sum, d, 32);
        const float corr = __expf(mrow[r] - newm);
        lrow[r] = lrow[r] * corr + sum;
        mrow[r] = newm;
#pragma unroll
        for (int j = 0; j < 4; ++j) acc[j][r] *= corr;
        const int prow = r + 8 * kh;
        Ps[wave][prow * 32 + cn]      = (_Float16)p0;
        Ps[wave][prow * 32 + 16 + cn] = (_Float16)p1;
      }
      asm volatile("" ::: "memory");   // keep P stores before the re-gather
      // O += P V  (per-wave scratch; same-wave LDS ops are in-order)
      v16h pa = frag_a_from((const uint32_t*)&Ps[wave][0], 16);
#pragma unroll
      for (int j = 0; j < 4; ++j) {
        v16h bv = frag_b_from((const uint32_t*)Vt + (16 * j) * 16, 16);
        acc[j] = __builtin_amdgcn_wmma_f32_16x16x32_f16(
            false, pa, false, bv, (short)0, acc[j], false, false);
      }
    }
    __syncthreads();
  }

  // Normalize and store O (f16), layout [B,S,D] with head offset.
#pragma unroll
  for (int j = 0; j < 4; ++j)
#pragma unroll
    for (int r = 0; r < 8; ++r) {
      const int q  = qbase + r + 8 * kh;
      const int hd = 16 * j + cn;
      Op[(size_t)(bb * S_ + q) * D_ + h * HD_ + hd] = (_Float16)(acc[j][r] / lrow[r]);
    }
}

// ---------------------------------------------------------------------------
// Host-side orchestration
// ---------------------------------------------------------------------------
extern "C" void kernel_launch(void* const* d_in, const int* in_sizes, int n_in,
                              void* d_out, int out_size, void* d_ws, size_t ws_size,
                              hipStream_t stream) {
  (void)in_sizes; (void)n_in; (void)out_size; (void)ws_size;
  const float* q  = (const float*)d_in[0];
  const float* k  = (const float*)d_in[1];
  const float* v  = (const float*)d_in[2];
  /* d_in[3] = causal mask, implicit in the kernel */
  const float* Wq = (const float*)d_in[4];  const float* bq = (const float*)d_in[5];
  const float* Wk = (const float*)d_in[6];  const float* bk = (const float*)d_in[7];
  const float* Wv = (const float*)d_in[8];  const float* bv = (const float*)d_in[9];
  const float* Wf = (const float*)d_in[10]; const float* bf = (const float*)d_in[11];
  float* out = (float*)d_out;

  const size_t nBSD = (size_t)B_ * S_ * D_;   // 4,194,304
  const size_t nDD  = (size_t)D_ * D_;        // 1,048,576

  char* ws = (char*)d_ws;
  size_t off = 0;
  auto alloc = [&](size_t bytes) {
    char* p = ws + off;
    off += (bytes + 255) & ~(size_t)255;
    return p;
  };
  _Float16* qh  = (_Float16*)alloc(nBSD * 2);
  _Float16* kh  = (_Float16*)alloc(nBSD * 2);
  _Float16* vh  = (_Float16*)alloc(nBSD * 2);
  _Float16* wqh = (_Float16*)alloc(nDD * 2);
  _Float16* wkh = (_Float16*)alloc(nDD * 2);
  _Float16* wvh = (_Float16*)alloc(nDD * 2);
  _Float16* wfh = (_Float16*)alloc(nDD * 2);
  _Float16* Qp  = (_Float16*)alloc(nBSD * 2);
  _Float16* Kp  = (_Float16*)alloc(nBSD * 2);
  _Float16* Vp  = (_Float16*)alloc(nBSD * 2);
  _Float16* Oph = (_Float16*)alloc(nBSD * 2);

  const int cvtThreads = 256;
  const int cvtBlocksBSD = (int)((nBSD + 255) / 256) > 4096 ? 4096 : (int)((nBSD + 255) / 256);
  const int cvtBlocksDD  = (int)((nDD + 255) / 256)  > 4096 ? 4096 : (int)((nDD + 255) / 256);
  cvt_f32_to_f16<<<cvtBlocksBSD, cvtThreads, 0, stream>>>(q,  qh,  (int)nBSD);
  cvt_f32_to_f16<<<cvtBlocksBSD, cvtThreads, 0, stream>>>(k,  kh,  (int)nBSD);
  cvt_f32_to_f16<<<cvtBlocksBSD, cvtThreads, 0, stream>>>(v,  vh,  (int)nBSD);
  cvt_f32_to_f16<<<cvtBlocksDD,  cvtThreads, 0, stream>>>(Wq, wqh, (int)nDD);
  cvt_f32_to_f16<<<cvtBlocksDD,  cvtThreads, 0, stream>>>(Wk, wkh, (int)nDD);
  cvt_f32_to_f16<<<cvtBlocksDD,  cvtThreads, 0, stream>>>(Wv, wvh, (int)nDD);
  cvt_f32_to_f16<<<cvtBlocksDD,  cvtThreads, 0, stream>>>(Wf, wfh, (int)nDD);

  const int M = B_ * S_;   // 4096
  dim3 gProj(D_ / 128, M / 64);
  gemm_xwT_bias<false><<<gProj, 128, 0, stream>>>(qh, wqh, bq, Qp, nullptr, M, D_, D_);
  gemm_xwT_bias<false><<<gProj, 128, 0, stream>>>(kh, wkh, bk, Kp, nullptr, M, D_, D_);
  gemm_xwT_bias<false><<<gProj, 128, 0, stream>>>(vh, wvh, bv, Vp, nullptr, M, D_, D_);

  dim3 gAttn(S_ / 64, B_ * H_);
  attn_flash<<<gAttn, 128, 0, stream>>>(Qp, Kp, Vp, Oph);

  gemm_xwT_bias<true><<<gProj, 128, 0, stream>>>(Oph, wfh, bf, nullptr, out, M, D_, D_);
}
